// SirenGINet_14250701488616
// MI455X (gfx1250) — matrix operand order repaired and checked by reading, not compiled, not run on hardware
//
#include <hip/hip_runtime.h>
#include <hip/hip_bf16.h>

typedef __attribute__((ext_vector_type(16))) _Float16     v16h;
typedef __attribute__((ext_vector_type(8)))  float        v8f;
typedef __attribute__((ext_vector_type(8)))  unsigned int v8u;

#define TABLE_SIZE 65536
#define WAVES      8
#define TILES      8
#define PIX_PER_BLOCK (WAVES * 16 * TILES)   // 1024

#define INV_2PI 0.15915494309189535f

union ABFrag { v16h h; v8u u; };

// Wave-local LDS ordering: activation tiles are private to a wave, and LDS ops
// from one wave complete in order (DScnt), so a dscnt-wait + compiler memory
// fence replaces block-wide barriers between layers.
__device__ inline void wave_lds_sync() {
  asm volatile("s_wait_dscnt 0" ::: "memory");
}

// sin(w0*x) with the w0 and 1/2pi scales fused into one constant; v_sin_f32
// takes its argument in rotations, so sin(w0*x) == v_sin_f32(x * w0/2pi).
__device__ inline float siren_sin(float x, float w0_over_2pi) {
  return __builtin_amdgcn_sinf(x * w0_over_2pi);
}

// ---- A fragment (16x32 f16, M x K): lane = M (0..15 twice), half = lane/16
// VGPR j: K = (j<4?0:16) + 8*half + 2*(j&3), pairs packed in 32-bit
__device__ inline v16h load_A_frag(const _Float16* act, int stride, int m, int k0, int half) {
  ABFrag a;
#pragma unroll
  for (int j = 0; j < 8; ++j) {
    int kb = k0 + ((j < 4) ? 0 : 16) + 8 * half + 2 * (j & 3);
    a.u[j] = *(const unsigned int*)(act + m * stride + kb);
  }
  return a.h;
}

// ---- B fragment (32x16 f16, K x N): lane%16 = N, VGPR j: K = (j<4?0:16) + 8*half + 2*(j&3)
// weights stored row-major [out][in_pad] so B[k][n] = W[n][k] -> contiguous pairs
__device__ inline v16h load_B_frag(const _Float16* w, int stride, int nrow, int k0, int half) {
  ABFrag b;
#pragma unroll
  for (int j = 0; j < 8; ++j) {
    int kb = k0 + ((j < 4) ? 0 : 16) + 8 * half + 2 * (j & 3);
    b.u[j] = *(const unsigned int*)(w + nrow * stride + kb);
  }
  return b.h;
}

// Degree-4 (16-component) real SH basis, matches reference constants
__device__ inline void sh_encode16(float x, float y, float z, float* o) {
  const float C0 = 0.28209479177387814f;
  const float C1 = 0.4886025119029199f;
  const float C20 = 1.0925484305920792f, C21 = -1.0925484305920792f;
  const float C22 = 0.31539156525252005f, C23 = -1.0925484305920792f;
  const float C24 = 0.5462742152960396f;
  const float C30 = -0.5900435899266435f, C31 = 2.890611442640554f;
  const float C32 = -0.4570457994644658f, C33 = 0.3731763325901154f;
  const float C34 = -0.4570457994644658f, C35 = 1.445305721320277f;
  const float C36 = -0.5900435899266435f;
  float xx = x * x, yy = y * y, zz = z * z;
  float xy = x * y, yz = y * z, xz = x * z;
  o[0] = C0;            o[1] = -C1 * y;       o[2] = C1 * z;        o[3] = -C1 * x;
  o[4] = C20 * xy;      o[5] = C21 * yz;      o[6] = C22 * (2.f * zz - xx - yy);
  o[7] = C23 * xz;      o[8] = C24 * (xx - yy);
  o[9]  = C30 * y * (3.f * xx - yy);
  o[10] = C31 * xy * z;
  o[11] = C32 * y * (4.f * zz - xx - yy);
  o[12] = C33 * z * (2.f * zz - 3.f * xx - 3.f * yy);
  o[13] = C34 * x * (4.f * zz - xx - yy);
  o[14] = C35 * z * (xx - yy);
  o[15] = C36 * x * (xx - 3.f * yy);
}

struct SmemLayout {
  _Float16 w_lm0[64 * 32];
  _Float16 w_lm1[64 * 64];
  _Float16 w_lm2[64 * 64];
  _Float16 w_lm3[16 * 64];
  _Float16 w_rf0[32 * 32];
  _Float16 w_rf1[32 * 32];
  _Float16 w_rf2[32 * 32];
  _Float16 w_rf3[16 * 32];
  float b_lm0[64], b_lm1[64], b_lm2[64], b_lm3[16];
  float b_rf0[32], b_rf1[32], b_rf2[32], b_rf3[16];
  _Float16 act[WAVES][2][16 * 64];   // per-wave ping-pong activation tiles
};

__device__ inline void coop_load_w(_Float16* dst, const float* src, int dout, int din,
                                   int dinp, int tid, int nth) {
  for (int i = tid; i < dout * dinp; i += nth) {
    int r = i / dinp, c = i % dinp;
    dst[i] = (c < din) ? (_Float16)src[r * din + c] : (_Float16)0.f;
  }
}
__device__ inline void coop_load_b(float* dst, const float* src, int n, int npad,
                                   int tid, int nth) {
  for (int i = tid; i < npad; i += nth) dst[i] = (i < n) ? src[i] : 0.f;
}

// One SIREN layer for a 16-pixel tile: D = sin(w0*(A*W^T + b)) via WMMA.
// A fragments hoisted into registers (loaded once, reused across N tiles);
// bias folded into the accumulator initializer; w0/2pi fused into one mul.
template <int KPAD, int NPAD>
__device__ inline void siren_layer_t(const _Float16* actIn, int sIn,
                                     const _Float16* W, const float* bias,
                                     _Float16* actOut, int sOut, int lane,
                                     float w0_over_2pi) {
  const int half = lane >> 4;
  const int nl   = lane & 15;
  v16h a[KPAD / 32];
#pragma unroll
  for (int ks = 0; ks < KPAD / 32; ++ks)
    a[ks] = load_A_frag(actIn, sIn, nl, ks * 32, half);
#pragma unroll
  for (int nt = 0; nt < NPAD / 16; ++nt) {
    const float bv = bias[nt * 16 + nl];
    v8f c = {bv, bv, bv, bv, bv, bv, bv, bv};
#pragma unroll
    for (int ks = 0; ks < KPAD / 32; ++ks) {
      v16h b = load_B_frag(W + nt * 16 * KPAD, KPAD, nl, ks * 32, half);
      c = __builtin_amdgcn_wmma_f32_16x16x32_f16(false, a[ks], false, b, (short)0, c,
                                                 false, false);
    }
#pragma unroll
    for (int r = 0; r < 8; ++r)
      actOut[(r + 8 * half) * sOut + nt * 16 + nl] =
          (_Float16)siren_sin(c[r], w0_over_2pi);
  }
}

__global__ void __launch_bounds__(256)
siren_ginet_kernel(const float* __restrict__ uv, const float* __restrict__ nrm,
                   const float* __restrict__ vvec, const float* __restrict__ tables,
                   const float* lw0, const float* lw1, const float* lw2, const float* lw3,
                   const float* lb0, const float* lb1, const float* lb2, const float* lb3,
                   const float* rw0, const float* rw1, const float* rw2, const float* rw3,
                   const float* rb0, const float* rb1, const float* rb2, const float* rb3,
                   float* __restrict__ out, int nPixels) {
  __shared__ SmemLayout sm;
  const int tid = threadIdx.x, nth = blockDim.x;

  // hint: keep hash tables warm in GL2 (global_prefetch_b8)
  if (tid < 4) __builtin_prefetch((const void*)(tables + (size_t)tid * TABLE_SIZE * 2), 0, 1);

  // stage weights (fp32 -> f16, K padded) and biases into LDS
  coop_load_w(sm.w_lm0, lw0, 64, 24, 32, tid, nth);
  coop_load_w(sm.w_lm1, lw1, 64, 64, 64, tid, nth);
  coop_load_w(sm.w_lm2, lw2, 64, 64, 64, tid, nth);
  coop_load_w(sm.w_lm3, lw3, 16, 64, 64, tid, nth);
  coop_load_w(sm.w_rf0, rw0, 32, 32, 32, tid, nth);
  coop_load_w(sm.w_rf1, rw1, 32, 32, 32, tid, nth);
  coop_load_w(sm.w_rf2, rw2, 32, 32, 32, tid, nth);
  coop_load_w(sm.w_rf3, rw3,  3, 32, 32, tid, nth);  // pad 3 -> 16 rows below
  for (int i = tid + 3 * 32; i < 16 * 32; i += nth) sm.w_rf3[i] = (_Float16)0.f;
  coop_load_b(sm.b_lm0, lb0, 64, 64, tid, nth);
  coop_load_b(sm.b_lm1, lb1, 64, 64, tid, nth);
  coop_load_b(sm.b_lm2, lb2, 64, 64, tid, nth);
  coop_load_b(sm.b_lm3, lb3, 16, 16, tid, nth);
  coop_load_b(sm.b_rf0, rb0, 32, 32, tid, nth);
  coop_load_b(sm.b_rf1, rb1, 32, 32, tid, nth);
  coop_load_b(sm.b_rf2, rb2, 32, 32, tid, nth);
  coop_load_b(sm.b_rf3, rb3,  3, 16, tid, nth);
  __syncthreads();  // only block-wide barrier needed: weights are shared

  const int wave = tid >> 5, lane = tid & 31;
  const int half = lane >> 4, m = lane & 15;
  const long waveBase = (long)blockIdx.x * PIX_PER_BLOCK + (long)wave * 16 * TILES;
  _Float16* A0 = sm.act[wave][0];
  _Float16* A1 = sm.act[wave][1];

#pragma unroll 1
  for (int t = 0; t < TILES; ++t) {
    const long base = waveBase + t * 16;
    long pix = base + m;
    if (pix >= nPixels) pix = nPixels - 1;  // clamp reads; writes guarded below

    float shv[16];
    if (lane < 16) {
      // --- SH(n) -> features 0..15
      float sh[16];
      sh_encode16(nrm[pix * 3 + 0], nrm[pix * 3 + 1], nrm[pix * 3 + 2], sh);
#pragma unroll
      for (int k = 0; k < 16; ++k) A0[m * 32 + k] = (_Float16)sh[k];
      // --- hashgrid(uv) -> features 16..23
      const float u0 = uv[pix * 2 + 0], u1 = uv[pix * 2 + 1];
#pragma unroll
      for (int lvl = 0; lvl < 4; ++lvl) {
        const float res = (float)(512 << lvl);
        float sx = u0 * res, sy = u1 * res;
        float fx = floorf(sx), fy = floorf(sy);
        unsigned int bx = (unsigned int)(int)fx, by = (unsigned int)(int)fy;
        float wx = sx - fx, wy = sy - fy;
        const float2* tab = (const float2*)(tables + (size_t)lvl * TABLE_SIZE * 2);
        float f0 = 0.f, f1 = 0.f;
#pragma unroll
        for (int c = 0; c < 4; ++c) {
          unsigned int cx = bx + (unsigned int)(c >> 1);
          unsigned int cy = by + (unsigned int)(c & 1);
          unsigned int h = (cx ^ (cy * 2654435761u)) & (TABLE_SIZE - 1);
          float2 tv = tab[h];
          float w = ((c >> 1) ? wx : 1.f - wx) * ((c & 1) ? wy : 1.f - wy);
          f0 += w * tv.x;
          f1 += w * tv.y;
        }
        A0[m * 32 + 16 + 2 * lvl + 0] = (_Float16)f0;
        A0[m * 32 + 16 + 2 * lvl + 1] = (_Float16)f1;
      }
#pragma unroll
      for (int k = 24; k < 32; ++k) A0[m * 32 + k] = (_Float16)0.f;  // pad 24->32
    } else {
      // lanes 16..31: SH(v) kept in registers until the rf concat
      sh_encode16(vvec[pix * 3 + 0], vvec[pix * 3 + 1], vvec[pix * 3 + 2], shv);
    }
    wave_lds_sync();

    // ---- lm MLP: 24(->32) -> 64 -> 64 -> 64 -> 16, sin activations
    siren_layer_t<32, 64>(A0, 32, sm.w_lm0, sm.b_lm0, A1, 64, lane, 20.f * INV_2PI);
    wave_lds_sync();
    siren_layer_t<64, 64>(A1, 64, sm.w_lm1, sm.b_lm1, A0, 64, lane, INV_2PI);
    wave_lds_sync();
    siren_layer_t<64, 64>(A0, 64, sm.w_lm2, sm.b_lm2, A1, 64, lane, INV_2PI);
    wave_lds_sync();
    siren_layer_t<64, 16>(A1, 64, sm.w_lm3, sm.b_lm3, A0, 32, lane, INV_2PI);  // cols 0..15
    if (lane >= 16) {  // concat SH(v) into cols 16..31 of rf input
#pragma unroll
      for (int k = 0; k < 16; ++k) A0[m * 32 + 16 + k] = (_Float16)shv[k];
    }
    wave_lds_sync();

    // ---- rf MLP: 32 -> 32 -> 32 -> 32 -> 3 (sigmoid)
    siren_layer_t<32, 32>(A0, 32, sm.w_rf0, sm.b_rf0, A1, 32, lane, INV_2PI);
    wave_lds_sync();
    siren_layer_t<32, 32>(A1, 32, sm.w_rf1, sm.b_rf1, A0, 32, lane, INV_2PI);
    wave_lds_sync();
    siren_layer_t<32, 32>(A0, 32, sm.w_rf2, sm.b_rf2, A1, 32, lane, INV_2PI);
    wave_lds_sync();

    {  // final 32 -> 3 with sigmoid, written straight to global
      const float bv = sm.b_rf3[m];
      v8f c = {bv, bv, bv, bv, bv, bv, bv, bv};
      v16h a = load_A_frag(A1, 32, m, 0, half);
      v16h b = load_B_frag(sm.w_rf3, 32, m, 0, half);
      c = __builtin_amdgcn_wmma_f32_16x16x32_f16(false, a, false, b, (short)0, c,
                                                 false, false);
      if (m < 3) {
#pragma unroll
        for (int r = 0; r < 8; ++r) {
          long prow = base + r + 8 * half;
          if (prow < nPixels)
            out[prow * 3 + m] = 1.f / (1.f + __expf(-c[r]));
        }
      }
    }
    wave_lds_sync();
  }
}

extern "C" void kernel_launch(void* const* d_in, const int* in_sizes, int n_in,
                              void* d_out, int out_size, void* d_ws, size_t ws_size,
                              hipStream_t stream) {
  const float* uv     = (const float*)d_in[0];
  const float* nrm    = (const float*)d_in[1];
  const float* vvec   = (const float*)d_in[2];
  const float* tables = (const float*)d_in[3];
  const float* lw[4]  = {(const float*)d_in[4],  (const float*)d_in[5],
                         (const float*)d_in[6],  (const float*)d_in[7]};
  const float* lb[4]  = {(const float*)d_in[8],  (const float*)d_in[9],
                         (const float*)d_in[10], (const float*)d_in[11]};
  const float* rw[4]  = {(const float*)d_in[12], (const float*)d_in[13],
                         (const float*)d_in[14], (const float*)d_in[15]};
  const float* rb[4]  = {(const float*)d_in[16], (const float*)d_in[17],
                         (const float*)d_in[18], (const float*)d_in[19]};
  float* out = (float*)d_out;

  const int nPixels = in_sizes[0] / 2;
  const int grid = (nPixels + PIX_PER_BLOCK - 1) / PIX_PER_BLOCK;
  siren_ginet_kernel<<<grid, 256, 0, stream>>>(
      uv, nrm, vvec, tables,
      lw[0], lw[1], lw[2], lw[3], lb[0], lb[1], lb[2], lb[3],
      rw[0], rw[1], rw[2], rw[3], rb[0], rb[1], rb[2], rb[3],
      out, nPixels);
}
